// Bezier_71150428225973
// MI455X (gfx1250) — matrix operand order repaired and checked by reading, not compiled, not run on hardware
//
#include <hip/hip_runtime.h>

// Bezier raster = A(8192x100) x B(100x8192) GEMM.
// 13.4 GFLOP, 256 MB output -> HBM store floor ~11us @ 23.3 TB/s.
// Use split-f16 (Markidis) WMMA: 3x v_wmma_f32_16x16x32_f16 per f32 product
// term -> ~1e-6 accuracy at f16-matrix throughput, memory-bound overall.

#define RES   8192
#define STEPS 100
#define KPAD  128          // K padded to 4 chunks of 32 (zeros beyond 99)
#define XK    136          // LDS row stride in f16 (4-bank lane stride, 16B aligned)
#define BT    128          // block output tile (128x128)

typedef __attribute__((ext_vector_type(8)))  _Float16 v8h;
typedef __attribute__((ext_vector_type(16))) _Float16 v16h;
typedef __attribute__((ext_vector_type(8)))  float    v8f;

union V16 { v16h v; v8h h[2]; };

__global__ __launch_bounds__(256) void bezier_raster_wmma(
    const float* __restrict__ cp, float* __restrict__ out)
{
    __shared__ __align__(16) _Float16 xsh[BT][XK];  // gx hi, row = out-row i, col = t
    __shared__ __align__(16) _Float16 xsl[BT][XK];  // gx lo
    __shared__ __align__(16) _Float16 ysh[BT][XK];  // gy hi, row = out-col j, col = t
    __shared__ __align__(16) _Float16 ysl[BT][XK];  // gy lo
    __shared__ float s_cx[STEPS];
    __shared__ float s_cy[STEPS];

    const int tid = threadIdx.x;
    const int bi  = blockIdx.y;   // output row block (x-dimension of curve)
    const int bj  = blockIdx.x;   // output col block (y-dimension of curve)

    // ---- cubic Bezier samples (De Casteljau), 100 threads ----
    if (tid < STEPS) {
        float t = (float)tid * (1.0f / (float)(STEPS - 1));
        float p0x = cp[0], p0y = cp[1];
        float p1x = cp[2], p1y = cp[3];
        float p2x = cp[4], p2y = cp[5];
        float p3x = cp[6], p3y = cp[7];
        float ax = p0x + (p1x - p0x) * t, ay = p0y + (p1y - p0y) * t;
        float bx = p1x + (p2x - p1x) * t, by = p1y + (p2y - p1y) * t;
        float cx = p2x + (p3x - p2x) * t, cy = p2y + (p3y - p2y) * t;
        float dx = ax + (bx - ax) * t,    dy = ay + (by - ay) * t;
        float ex = bx + (cx - bx) * t,    ey = by + (cy - by) * t;
        s_cx[tid] = dx + (ex - dx) * t;
        s_cy[tid] = dy + (ey - dy) * t;
    }
    __syncthreads();

    // ---- fill LDS: gaussian strips in split-f16 (hi + lo) ----
    for (int idx = tid; idx < BT * KPAD; idx += 256) {
        int i = idx >> 7;          // pixel within tile
        int t = idx & (KPAD - 1);  // K index
        float gx = 0.0f, gy = 0.0f;
        if (t < STEPS) {
            float x  = (float)(bi * BT + i) * (1.0f / (float)RES);
            float y  = (float)(bj * BT + i) * (1.0f / (float)RES);
            float ddx = s_cx[t] - x;
            float ddy = s_cy[t] - y;
            gx = __expf(ddx * ddx * -5000.0f);   // 1/0.0002 = 5000
            gy = __expf(ddy * ddy * -5000.0f);
        }
        _Float16 hx = (_Float16)gx;
        _Float16 hy = (_Float16)gy;
        xsh[i][t] = hx;  xsl[i][t] = (_Float16)(gx - (float)hx);
        ysh[i][t] = hy;  ysl[i][t] = (_Float16)(gy - (float)hy);
    }
    __syncthreads();

    // ---- WMMA: each wave owns a 32x64 sub-tile (2x4 tiles of 16x16) ----
    const int lane = tid & 31;
    const int wid  = tid >> 5;
    const int lo   = lane & 15;
    const int hh   = lane >> 4;        // lane half: 0 (lanes 0-15) / 1 (16-31)
    const int wm   = (wid >> 1) * 32;  // wave row origin within tile
    const int wn   = (wid & 1) * 64;   // wave col origin within tile

    v8f acc[2][4] = {};

    for (int k0 = 0; k0 < KPAD; k0 += 32) {
        // A fragments (16x32 f16): K runs of 8 at k0+8*hh and k0+16+8*hh, M=lo
        V16 ah[2], al[2];
        #pragma unroll
        for (int r = 0; r < 2; ++r) {
            const _Float16* rh = &xsh[wm + 16 * r + lo][0];
            const _Float16* rl = &xsl[wm + 16 * r + lo][0];
            ah[r].h[0] = *(const v8h*)(rh + k0 + 8 * hh);
            ah[r].h[1] = *(const v8h*)(rh + k0 + 16 + 8 * hh);
            al[r].h[0] = *(const v8h*)(rl + k0 + 8 * hh);
            al[r].h[1] = *(const v8h*)(rl + k0 + 16 + 8 * hh);
        }
        // B fragments (32x16 f16): K runs of 8 at k0+8*hh and k0+16+8*hh, N=lo
        V16 bh[4], bl[4];
        #pragma unroll
        for (int c = 0; c < 4; ++c) {
            const _Float16* rh = &ysh[wn + 16 * c + lo][0];
            const _Float16* rl = &ysl[wn + 16 * c + lo][0];
            bh[c].h[0] = *(const v8h*)(rh + k0 + 8 * hh);
            bh[c].h[1] = *(const v8h*)(rh + k0 + 16 + 8 * hh);
            bl[c].h[0] = *(const v8h*)(rl + k0 + 8 * hh);
            bl[c].h[1] = *(const v8h*)(rl + k0 + 16 + 8 * hh);
        }
        // Split product: hi*hi + hi*lo + lo*hi (lo*lo ~2^-24, dropped)
        #pragma unroll
        for (int r = 0; r < 2; ++r) {
            #pragma unroll
            for (int c = 0; c < 4; ++c) {
                acc[r][c] = __builtin_amdgcn_wmma_f32_16x16x32_f16(
                    false, ah[r].v, false, bh[c].v, (short)0, acc[r][c], false, false);
                acc[r][c] = __builtin_amdgcn_wmma_f32_16x16x32_f16(
                    false, ah[r].v, false, bl[c].v, (short)0, acc[r][c], false, false);
                acc[r][c] = __builtin_amdgcn_wmma_f32_16x16x32_f16(
                    false, al[r].v, false, bh[c].v, (short)0, acc[r][c], false, false);
            }
        }
    }

    // ---- store: C/D layout -> row = 8*half + vgpr, col = lane%16 ----
    const float scale = 1.0f / (float)STEPS;
    #pragma unroll
    for (int r = 0; r < 2; ++r) {
        #pragma unroll
        for (int c = 0; c < 4; ++c) {
            size_t col  = (size_t)bj * BT + wn + 16 * c + lo;
            size_t row0 = (size_t)bi * BT + wm + 16 * r + 8 * hh;
            float* p = out + row0 * (size_t)RES + col;
            #pragma unroll
            for (int v = 0; v < 8; ++v)
                __builtin_nontemporal_store(acc[r][c][v] * scale,
                                            p + (size_t)v * (size_t)RES);
        }
    }
}

extern "C" void kernel_launch(void* const* d_in, const int* in_sizes, int n_in,
                              void* d_out, int out_size, void* d_ws, size_t ws_size,
                              hipStream_t stream) {
    const float* cp  = (const float*)d_in[0];   // control_points: (4,2) f32
    float*       out = (float*)d_out;           // (8192, 8192) f32
    dim3 grid(RES / BT, RES / BT);
    bezier_raster_wmma<<<grid, dim3(256), 0, stream>>>(cp, out);
}
